// FusionModel_79963701117412
// MI455X (gfx1250) — matrix operand order repaired and checked by reading, not compiled, not run on hardware
//
#include <hip/hip_runtime.h>

// ---------------------------------------------------------------------------
// Model dims (match reference)
// ---------------------------------------------------------------------------
#define BATCH   512
#define TSEQ    128
#define HID     128
#define EMB     256          // 2*H
#define NHEAD   8
#define HD      32           // EMB/NHEAD
#define BT      (BATCH * TSEQ)
#define LOG2PI_F 1.8378770664093453f

typedef __attribute__((ext_vector_type(16))) _Float16 v16h;
typedef __attribute__((ext_vector_type(8)))  _Float16 v8h;
typedef __attribute__((ext_vector_type(8)))  float    v8f;

// ---------------------------------------------------------------------------
// v_wmma_f32_16x16x32_f16 fragment maps (CDNA5 ISA):
//   A (16x32): lane m = lane&15, hi = lane>>4; halfs e=0..7 -> K = hi*8+e,
//              halfs e=8..15 -> K = 16+hi*8+(e-8)  => two contiguous 8-half runs
//   B (32x16): lane n = lane&15, hi; halfs e -> same K map as A (two 8-half runs)
//   C/D f32  : vgpr r, lane -> M = r + hi*8, N = lane&15
// Fragments load as two 16-byte LDS chunks (ds_load_b128).
// ---------------------------------------------------------------------------
__device__ __forceinline__ v16h frag_ld(const _Float16* p0, const _Float16* p1) {
    v8h x = *(const v8h*)p0;
    v8h y = *(const v8h*)p1;
    return __builtin_shufflevector(x, y, 0, 1, 2, 3, 4, 5, 6, 7,
                                         8, 9, 10, 11, 12, 13, 14, 15);
}
__device__ __forceinline__ v8f wmma_f16(v16h a, v16h b, v8f c) {
    return __builtin_amdgcn_wmma_f32_16x16x32_f16(false, a, false, b, (short)0, c,
                                                  false, false);
}

// ---------------------------------------------------------------------------
// Aligned GEMM: C[m, col_off+n] = act( A[m,:K] . W[n,:K] + bias[n] )
// REQUIRES: gridDim covers M,N exactly (M%64==0, N%64==0) and K%32==0.
// 256 threads / 8 waves per 64x64 tile; LDS-staged f16 panels; each wave
// computes two 16x16 subtiles sharing one B fragment. No bounds checks ->
// no exec-mask thrash. Next K-panel is prefetched (global_prefetch_b8).
// act==1: leaky_relu(0.01); bn_g: folded BN after leaky.
// ---------------------------------------------------------------------------
__global__ void __launch_bounds__(256)
wmma_gemm_kernel(const float* __restrict__ A, int lda,
                 const float* __restrict__ W, int ldw,
                 const float* __restrict__ bias,
                 float* __restrict__ C, long ldc, int col_off,
                 int K, int act,
                 const float* __restrict__ bn_g, const float* __restrict__ bn_b,
                 const float* __restrict__ bn_m, const float* __restrict__ bn_v) {
    __shared__ _Float16 As[64 * 32];
    __shared__ _Float16 Ws[64 * 32];
    const int tid  = threadIdx.x;
    const int wave = tid >> 5, lane = tid & 31;
    const int hi   = lane >> 4, ln = lane & 15;
    const int m0   = blockIdx.x * 64, n0 = blockIdx.y * 64;
    const int ni   = wave & 3;            // n-subtile of this wave
    const int mi0  = (wave >> 2) << 1;    // first of two m-subtiles
    const int sr   = tid >> 2;            // staging row 0..63
    const int sc   = (tid & 3) * 8;       // staging col 0/8/16/24

    v8f acc0 = {}, acc1 = {};
    for (int k0 = 0; k0 < K; k0 += 32) {
        if (k0) __syncthreads();
        const float* ap = A + (long)(m0 + sr) * lda + k0 + sc;
        const float* wp = W + (long)(n0 + sr) * ldw + k0 + sc;
        if (k0 + 32 < K) {                      // warm next panel in cache
            __builtin_prefetch(ap + 32, 0, 1);  // -> global_prefetch_b8
            __builtin_prefetch(wp + 32, 0, 1);
        }
#pragma unroll
        for (int i = 0; i < 8; ++i) {
            As[sr * 32 + sc + i] = (_Float16)ap[i];
            Ws[sr * 32 + sc + i] = (_Float16)wp[i];
        }
        __syncthreads();
        const int wrow = (ni * 16 + ln) * 32;
        v16h b  = frag_ld(&Ws[wrow + hi * 8], &Ws[wrow + 16 + hi * 8]);
        const int ar0 = (mi0 * 16 + ln) * 32;
        const int ar1 = ((mi0 + 1) * 16 + ln) * 32;
        v16h a0 = frag_ld(&As[ar0 + hi * 8], &As[ar0 + 16 + hi * 8]);
        v16h a1 = frag_ld(&As[ar1 + hi * 8], &As[ar1 + 16 + hi * 8]);
        acc0 = wmma_f16(a0, b, acc0);
        acc1 = wmma_f16(a1, b, acc1);
    }

    const int n = n0 + ni * 16 + ln;
    float bv = bias ? bias[n] : 0.0f;
    float scale = 1.0f, shift = 0.0f;
    if (bn_g) {
        scale = bn_g[n] * rsqrtf(bn_v[n] + 1e-5f);
        shift = bn_b[n] - bn_m[n] * scale;
    }
#pragma unroll
    for (int s = 0; s < 2; ++s) {
        v8f acc = s ? acc1 : acc0;
        const int mb = m0 + (mi0 + s) * 16 + hi * 8;
#pragma unroll
        for (int r = 0; r < 8; ++r) {
            float v = acc[r] + bv;
            if (act) v = (v > 0.0f) ? v : 0.01f * v;
            if (bn_g) v = v * scale + shift;
            C[(long)(mb + r) * ldc + col_off + n] = v;
        }
    }
}

// ---------------------------------------------------------------------------
// Branch-free zero-pad: dst(rows, dcols) <- src(rows, scols), zeros beyond.
// ---------------------------------------------------------------------------
__global__ void __launch_bounds__(256)
pad_kernel(const float* __restrict__ src, int rows, int scols,
           float* __restrict__ dst, int dcols) {
    const int i = blockIdx.x * 256 + threadIdx.x;
    if (i >= rows * dcols) return;
    const int r = i / dcols, c = i % dcols;
    const float v = src[(long)r * scols + (c < scols ? c : 0)];
    dst[i] = (c < scols) ? v : 0.0f;
}

// ---------------------------------------------------------------------------
// GRU scan: full 128-step recurrence per kernel. Workgroup owns 16 batch rows.
// Whh pre-converted to f16 resident in LDS (98 KB of the 320 KB/WGP pool),
// h kept as f32 + f16 mirror, so all per-step fragments are ds_load_b128.
// gh = h @ Whh^T + bhh : M=16, N=384, K=128 -> 24 tiles over 8 waves.
// The recurrence is the latency-critical chain -> prefetch next step's Gi
// lines during the current gate update so the next matmul phase covers HBM.
// ---------------------------------------------------------------------------
#define GRU_MB 16
#define GRU_SMEM (98304 + 8192 + 4096 + 24576)
__global__ void __launch_bounds__(256)
gru_scan_kernel(const float* __restrict__ Gi,   // (B, T, 384)
                const float* __restrict__ Whh,  // (384, 128)
                const float* __restrict__ bhh,  // (384)
                float* __restrict__ Y, int ldy, int col_off,
                int T, int reverse) {
    extern __shared__ char smem[];
    _Float16* whh16 = (_Float16*)smem;                           // 384*128 f16
    float*    h32   = (float*)(smem + 98304);                    // 16*128 f32
    _Float16* h16   = (_Float16*)(smem + 98304 + 8192);          // 16*128 f16
    float*    gh    = (float*)(smem + 98304 + 8192 + 4096);      // 16*384 f32
    const int b0   = blockIdx.x * GRU_MB;
    const int tid  = threadIdx.x;
    const int wave = tid >> 5, lane = tid & 31;
    const int hi   = lane >> 4, ln = lane & 15;

    for (int i = tid; i < 384 * HID; i += 256) whh16[i] = (_Float16)Whh[i];
    for (int i = tid; i < GRU_MB * HID; i += 256) {
        h32[i] = 0.0f;
        h16[i] = (_Float16)0.0f;
    }
    __syncthreads();

    for (int step = 0; step < T; ++step) {
        const int t = reverse ? (T - 1 - step) : step;

        for (int nt = wave; nt < 24; nt += 8) {
            const int n = nt * 16 + ln;
            v8f acc = {};
#pragma unroll
            for (int k0 = 0; k0 < HID; k0 += 32) {
                const int arow = ln * HID + k0;
                const int brow = n * HID + k0;
                v16h a = frag_ld(&h16[arow + hi * 8], &h16[arow + 16 + hi * 8]);
                v16h b = frag_ld(&whh16[brow + hi * 8], &whh16[brow + 16 + hi * 8]);
                acc = wmma_f16(a, b, acc);
            }
            const float bv = bhh[n];
#pragma unroll
            for (int r = 0; r < 8; ++r)
                gh[(r + hi * 8) * 384 + n] = acc[r] + bv;
        }
        __syncthreads();

        const bool more = (step + 1 < T);
        const int  t2   = reverse ? (t - 1) : (t + 1);
        for (int i = tid; i < GRU_MB * HID; i += 256) {
            const int m = i >> 7, j = i & 127;
            const long gib = ((long)(b0 + m) * T + t) * 384;
            if (more) {                         // warm next step's Gi lines
                const long gib2 = ((long)(b0 + m) * T + t2) * 384;
                __builtin_prefetch(Gi + gib2 + j, 0, 1);
                __builtin_prefetch(Gi + gib2 + HID + j, 0, 1);
                __builtin_prefetch(Gi + gib2 + 2 * HID + j, 0, 1);
            }
            const float gir = Gi[gib + j];
            const float giz = Gi[gib + HID + j];
            const float gin = Gi[gib + 2 * HID + j];
            const float ghr = gh[m * 384 + j];
            const float ghz = gh[m * 384 + HID + j];
            const float ghn = gh[m * 384 + 2 * HID + j];
            const float r  = 1.0f / (1.0f + __expf(-(gir + ghr)));
            const float z  = 1.0f / (1.0f + __expf(-(giz + ghz)));
            const float nn = tanhf(gin + r * ghn);
            const float hnew = (1.0f - z) * nn + z * h32[i];
            h32[i] = hnew;
            h16[i] = (_Float16)hnew;
            Y[((long)(b0 + m) * T + t) * ldy + col_off + j] = hnew;
        }
        __syncthreads();
    }
}

// ---------------------------------------------------------------------------
// Attention per (batch, head): QK^T/sqrt(32) -> softmax -> P @ V, all WMMA.
// q/k/v^T staged once as f16 in LDS; probabilities re-stored as f16 so the
// P@V phase is also pure b128 fragment loads. 120 KB dynamic LDS.
// ---------------------------------------------------------------------------
#define ATTN_SMEM (8192 + 8192 + 8192 + 65536 + 32768)
__global__ void __launch_bounds__(256)
attn_kernel(const float* __restrict__ qkv,   // (B, T, 768)
            float* __restrict__ attout) {    // (B, T, 256)
    extern __shared__ char smem[];
    _Float16* q16 = (_Float16*)smem;                   // 128*32
    _Float16* k16 = (_Float16*)(smem + 8192);          // 128*32
    _Float16* vt  = (_Float16*)(smem + 16384);         // 32*128 (transposed)
    float*    s   = (float*)(smem + 24576);            // 128*128 f32
    _Float16* p16 = (_Float16*)(smem + 24576 + 65536); // 128*128 f16
    const int b    = blockIdx.x >> 3;
    const int h    = blockIdx.x & 7;
    const int tid  = threadIdx.x;
    const int wave = tid >> 5, lane = tid & 31;
    const int hi   = lane >> 4, ln = lane & 15;

    const float* qb = qkv + (long)b * TSEQ * (3 * EMB) + h * HD;
    const float* kb = qb + EMB;
    const float* vb = qb + 2 * EMB;
    const float scl = 0.17677669529663687f;   // 1/sqrt(32)

    for (int idx = tid; idx < TSEQ * HD; idx += 256) {
        const int t = idx >> 5, d = idx & 31;
        q16[idx]         = (_Float16)qb[(long)t * (3 * EMB) + d];
        k16[idx]         = (_Float16)kb[(long)t * (3 * EMB) + d];
        vt[d * TSEQ + t] = (_Float16)vb[(long)t * (3 * EMB) + d];
    }
    __syncthreads();

    // Phase 1: scores (K = 32, single wmma step). 64 tiles / 8 waves.
    for (int tile = wave; tile < 64; tile += 8) {
        const int i0 = (tile >> 3) * 16, j0 = (tile & 7) * 16;
        const int ar = (i0 + ln) * HD;
        const int br = (j0 + ln) * HD;
        v16h a  = frag_ld(&q16[ar + hi * 8], &q16[ar + 16 + hi * 8]);
        v16h bb = frag_ld(&k16[br + hi * 8], &k16[br + 16 + hi * 8]);
        v8f acc = {};
        acc = wmma_f16(a, bb, acc);
#pragma unroll
        for (int r = 0; r < 8; ++r)
            s[(i0 + r + hi * 8) * TSEQ + j0 + ln] = acc[r] * scl;
    }
    __syncthreads();

    // Phase 2: row softmax, f32 in LDS; emit probabilities as f16.
    if (tid < TSEQ) {
        float mx = -1e30f;
        for (int j = 0; j < TSEQ; ++j) mx = fmaxf(mx, s[tid * TSEQ + j]);
        float sum = 0.0f;
        for (int j = 0; j < TSEQ; ++j) {
            const float e = __expf(s[tid * TSEQ + j] - mx);
            s[tid * TSEQ + j] = e;
            sum += e;
        }
        const float inv = 1.0f / sum;
        for (int j = 0; j < TSEQ; ++j)
            p16[tid * TSEQ + j] = (_Float16)(s[tid * TSEQ + j] * inv);
    }
    __syncthreads();

    // Phase 3: O = P @ V (M=128, K=128, N=32). 16 tiles / 8 waves.
    for (int tile = wave; tile < 16; tile += 8) {
        const int i0 = (tile >> 1) * 16, n0 = (tile & 1) * 16;
        v8f acc = {};
#pragma unroll
        for (int k0 = 0; k0 < TSEQ; k0 += 32) {
            const int ar = (i0 + ln) * TSEQ + k0;
            const int br = (n0 + ln) * TSEQ + k0;
            v16h a  = frag_ld(&p16[ar + hi * 8], &p16[ar + 16 + hi * 8]);
            v16h bb = frag_ld(&vt[br + hi * 8], &vt[br + 16 + hi * 8]);
            acc = wmma_f16(a, bb, acc);
        }
#pragma unroll
        for (int r = 0; r < 8; ++r)
            attout[((long)b * TSEQ + i0 + r + hi * 8) * EMB + h * HD + n0 + ln] = acc[r];
    }
}

// ---------------------------------------------------------------------------
// Mean over T (output projection hoisted past the mean: mean(X)@W^T + b).
// ---------------------------------------------------------------------------
__global__ void __launch_bounds__(256)
mean_kernel(const float* __restrict__ attout, float* __restrict__ out) {
    const int i = blockIdx.x * blockDim.x + threadIdx.x;
    if (i >= BATCH * EMB) return;
    const int b = i >> 8, e = i & 255;
    float sacc = 0.0f;
    for (int t = 0; t < TSEQ; ++t)
        sacc += attout[((long)b * TSEQ + t) * EMB + e];
    out[i] = sacc * (1.0f / TSEQ);
}

// ---------------------------------------------------------------------------
// user/touch features: fin[:,512:640] = cond[:,3:35]@Wu^T + bu ;
//                      fin[:,640:768] = cond[:,0:3]@Wt^T + bt.
// ---------------------------------------------------------------------------
__global__ void __launch_bounds__(256)
feat_ut_kernel(const float* __restrict__ cond,
               const float* __restrict__ Wu, const float* __restrict__ bu,
               const float* __restrict__ Wt, const float* __restrict__ bt,
               float* __restrict__ fin) {
    const int i = blockIdx.x * 256 + threadIdx.x;
    if (i >= BATCH * 128) return;
    const int b = i >> 7, j = i & 127;
    const float* cu = cond + (long)b * 35 + 3;
    float su = bu[j];
#pragma unroll
    for (int k = 0; k < 32; ++k) su += cu[k] * Wu[j * 32 + k];
    const float* ct = cond + (long)b * 35;
    float st = bt[j] + ct[0] * Wt[j * 3 + 0] + ct[1] * Wt[j * 3 + 1]
             + ct[2] * Wt[j * 3 + 2];
    fin[(long)b * 768 + 512 + j] = su;
    fin[(long)b * 768 + 640 + j] = st;
}

// ---------------------------------------------------------------------------
// logits = h2 @ w3^T + b3  (N = 8: straight-line dots, avoids padded WMMA).
// ---------------------------------------------------------------------------
__global__ void __launch_bounds__(256)
logits_kernel(const float* __restrict__ h2, const float* __restrict__ w3,
              const float* __restrict__ b3, float* __restrict__ lg) {
    const int i = blockIdx.x * 256 + threadIdx.x;
    if (i >= BATCH * 8) return;
    const int b = i >> 3, e = i & 7;
    float s = b3[e];
#pragma unroll 8
    for (int k = 0; k < 128; ++k) s += h2[(long)b * 128 + k] * w3[e * 128 + k];
    lg[i] = s;
}

// ---------------------------------------------------------------------------
// Mixture density tail.
// ---------------------------------------------------------------------------
__global__ void __launch_bounds__(256)
mixture_kernel(const float* __restrict__ logits,      // (B, 8)
               const float* __restrict__ temperature, // scalar
               const float* __restrict__ paras,       // (8, 13)
               const float* __restrict__ condition,   // (B, 35)
               const float* __restrict__ touch_xy,    // (B, 16, 2)
               float* __restrict__ out) {             // (B, 16)
    const int i = blockIdx.x * blockDim.x + threadIdx.x;
    if (i >= BATCH * 16) return;
    const int b = i >> 4, tgt = i & 15;
    const float invT = 1.0f / temperature[0];

    float lw[8];
    float mx = -1e30f;
#pragma unroll
    for (int e = 0; e < 8; ++e) {
        lw[e] = logits[b * 8 + e] * invT;
        mx = fmaxf(mx, lw[e]);
    }
    float sum = 0.0f;
#pragma unroll
    for (int e = 0; e < 8; ++e) sum += __expf(lw[e] - mx);
    const float lse = mx + __logf(sum);

    const float x1 = condition[(long)b * 35 + 0];
    const float x2 = condition[(long)b * 35 + 1];
    const float x3 = condition[(long)b * 35 + 2];
    const float tx = touch_xy[((long)b * 16 + tgt) * 2 + 0];
    const float ty = touch_xy[((long)b * 16 + tgt) * 2 + 1];

    float vals[8];
    float m2 = -1e30f;
#pragma unroll
    for (int e = 0; e < 8; ++e) {
        const float* p = paras + e * 13;
        const float mu = p[0] + p[1] * x2 + p[2] * x1 + p[3] * x3;
        const float sx = sqrtf(fabsf(p[4]) + fabsf(p[5]) * x2 * x2 +
                               fabsf(p[6]) * x1 * x1 +
                               fabsf(p[7]) * (x2 / (x1 + 1e-6f)) +
                               fabsf(p[8]) * x3 * x3 + 1e-6f);
        const float sy = sqrtf(fabsf(p[9]) + fabsf(p[10]) * x2 * x2 +
                               fabsf(p[11]) * x1 * x1 +
                               fabsf(p[12]) * x3 * x3 + 1e-6f);
        const float dx = (tx - mu) / sx;
        const float dy = ty / sy;
        const float clp = -LOG2PI_F - __logf(sx) - __logf(sy)
                          - 0.5f * (dx * dx + dy * dy);
        vals[e] = (lw[e] - lse) + clp;
        m2 = fmaxf(m2, vals[e]);
    }
    float s2 = 0.0f;
#pragma unroll
    for (int e = 0; e < 8; ++e) s2 += __expf(vals[e] - m2);
    out[i] = m2 + __logf(s2);
}

// ---------------------------------------------------------------------------
// Host orchestration
// ---------------------------------------------------------------------------
static void launch_gemm(const float* A, int lda, const float* W, int ldw,
                        const float* bias, float* C, long ldc, int col_off,
                        int M, int N, int K, int act, hipStream_t stream,
                        const float* g = nullptr, const float* bb = nullptr,
                        const float* mm = nullptr, const float* vv = nullptr) {
    dim3 grid(M / 64, N / 64);   // all call sites are 64/32-aligned
    wmma_gemm_kernel<<<grid, dim3(256), 0, stream>>>(A, lda, W, ldw, bias, C, ldc,
                                                     col_off, K, act, g, bb, mm, vv);
}

extern "C" void kernel_launch(void* const* d_in, const int* in_sizes, int n_in,
                              void* d_out, int out_size, void* d_ws, size_t ws_size,
                              hipStream_t stream) {
    (void)in_sizes; (void)n_in; (void)out_size; (void)ws_size;
    auto F = [&](int i) { return (const float*)d_in[i]; };

    // Input indices (setup_inputs dict order, dicts flattened in insertion order)
    // 0 condition  1 touch_xy  2 vib  3 acc  4 adapted_paras  5 temperature
    // 6 W_user 7 b_user 8 W_touch 9 b_touch
    // 10..25 vib_gru {l0:[Wih_f,Whh_f,bih_f,bhh_f,Wih_b,Whh_b,bih_b,bhh_b], l1:...}
    // 26..41 acc_gru      42..45 vib_attn {in_w,in_b,out_w,out_b}  46..49 acc_attn
    // 50..59 fus {w1,b1,bn_g,bn_b,bn_m,bn_v,w2,b2,w3,b3}

    float* ws       = (float*)d_ws;
    float* buf_gi   = ws;                            // BT*384
    float* buf_x1   = buf_gi   + (long)BT * 384;     // BT*256
    float* buf_x2   = buf_x1   + (long)BT * 256;     // BT*256
    float* buf_qkv  = buf_x2   + (long)BT * 256;     // BT*768
    float* buf_att  = buf_qkv  + (long)BT * 768;     // BT*256
    float* buf_mean = buf_att  + (long)BT * 256;     // B*256
    float* buf_fin  = buf_mean + (long)BATCH * 256;  // B*768
    float* buf_h1   = buf_fin  + (long)BATCH * 768;  // B*256
    float* buf_h2   = buf_h1   + (long)BATCH * 256;  // B*128
    float* buf_lg   = buf_h2   + (long)BATCH * 128;  // B*8 (pad 16)
    float* buf_xpad = buf_lg   + (long)BATCH * 16;   // BT*32
    float* buf_wpad = buf_xpad + (long)BT * 32;      // 384*32

    for (int g = 0; g < 2; ++g) {
        const int    base = (g == 0) ? 10 : 26;
        const float* x    = F((g == 0) ? 2 : 3);
        const int    K0   = (g == 0) ? 12 : 3;

        // zero-pad the raw input to K=32 so every GEMM hits the aligned path
        pad_kernel<<<(BT * 32 + 255) / 256, 256, 0, stream>>>(x, BT, K0,
                                                              buf_xpad, 32);
        for (int l = 0; l < 2; ++l) {
            float* Y = (l == 0) ? buf_x1 : buf_x2;
            for (int dir = 0; dir < 2; ++dir) {
                const int o = base + l * 8 + dir * 4;
                if (l == 0) {
                    // pad Wih (384 x K0) to (384 x 32)
                    pad_kernel<<<(384 * 32 + 255) / 256, 256, 0, stream>>>(
                        F(o + 0), 384, K0, buf_wpad, 32);
                    launch_gemm(buf_xpad, 32, buf_wpad, 32, F(o + 2),
                                buf_gi, 384, 0, BT, 384, 32, 0, stream);
                } else {
                    launch_gemm(buf_x1, 256, F(o + 0), 256, F(o + 2),
                                buf_gi, 384, 0, BT, 384, 256, 0, stream);
                }
                // 128-step recurrence, Whh + state resident in LDS
                gru_scan_kernel<<<BATCH / GRU_MB, 256, GRU_SMEM, stream>>>(
                    buf_gi, F(o + 1), F(o + 3), Y, 256, dir * HID, TSEQ, dir);
            }
        }

        const int ab = (g == 0) ? 42 : 46;
        launch_gemm(buf_x2, 256, F(ab + 0), 256, F(ab + 1),
                    buf_qkv, 768, 0, BT, 768, 256, 0, stream);
        attn_kernel<<<BATCH * NHEAD, 256, ATTN_SMEM, stream>>>(buf_qkv, buf_att);
        mean_kernel<<<(BATCH * EMB) / 256, 256, 0, stream>>>(buf_att, buf_mean);
        // mean(X @ Wout^T + b) == mean(X) @ Wout^T + b  -> 512-row GEMM
        launch_gemm(buf_mean, 256, F(ab + 2), 256, F(ab + 3),
                    buf_fin, 768, g * 256, BATCH, 256, 256, 0, stream);
    }

    feat_ut_kernel<<<(BATCH * 128 + 255) / 256, 256, 0, stream>>>(
        F(0), F(6), F(7), F(8), F(9), buf_fin);

    launch_gemm(buf_fin, 768, F(50), 768, F(51), buf_h1, 256, 0,
                BATCH, 256, 768, 1, stream, F(52), F(53), F(54), F(55));
    launch_gemm(buf_h1, 256, F(56), 256, F(57), buf_h2, 128, 0,
                BATCH, 128, 256, 1, stream);
    logits_kernel<<<(BATCH * 8 + 255) / 256, 256, 0, stream>>>(
        buf_h2, F(58), F(59), buf_lg);

    mixture_kernel<<<(BATCH * 16 + 255) / 256, 256, 0, stream>>>(
        buf_lg, F(5), F(4), F(0), F(1), (float*)d_out);
}
